// jointBert_82454782148801
// MI455X (gfx1250) — matrix operand order repaired and checked by reading, not compiled, not run on hardware
//
#include <hip/hip_runtime.h>
#include <hip/hip_bf16.h>

// ---------------- model dims ----------------
#define Bsz   64
#define Ssz   128
#define Hsz   768
#define Lnum  6
#define NHn   12
#define DHn   64
#define FFn   3072
#define NINT  22
#define NTAG  122
#define TOKS  (Bsz*Ssz)          // 8192
#define LN_EPS 1e-12f
#define INV_SQRT_DH 0.125f       // 1/sqrt(64)

// ---------------- WMMA vector types ----------------
typedef __attribute__((ext_vector_type(16))) __bf16         v16bf;
typedef __attribute__((ext_vector_type(16))) unsigned short v16u;
typedef __attribute__((ext_vector_type(8)))  float          v8f;

__device__ inline v8f zero8(){ v8f z; for (int i=0;i<8;i++) z[i]=0.f; return z; }

__device__ inline unsigned short f2bfu(float f){
    union { float f; unsigned u; } x; x.f = f;
    unsigned r = (x.u + 0x7FFFu + ((x.u >> 16) & 1u)) >> 16;   // RNE
    return (unsigned short)r;
}
__device__ inline float bfu2f(unsigned short s){
    union { unsigned u; float f; } x; x.u = ((unsigned)s) << 16; return x.f;
}
__device__ inline float gelu_f(float x){ return 0.5f*x*(1.0f+erff(x*0.70710678118f)); }

__device__ inline v8f wmma_bf16(v16u a, v16u b, v8f c){
    v16bf af = __builtin_bit_cast(v16bf, a);
    v16bf bf = __builtin_bit_cast(v16bf, b);
    return __builtin_amdgcn_wmma_f32_16x16x32_bf16(false, af, false, bf,
                                                   (short)0, c, false, false);
}

// Load one 16-element bf16 fragment as two contiguous 16B LDS chunks.
// Requires (row base + off)*2 to be 16B aligned.
__device__ inline v16u frag2(const unsigned short* __restrict__ row, int off0, int off1){
    struct P { uint4 a, b; } u;
    u.a = *(const uint4*)(row + off0);
    u.b = *(const uint4*)(row + off1);
    return __builtin_bit_cast(v16u, u);
}

// ============================================================
// GEMM:  C[M,N] = ACT(A[M,K](bf16) * B[K,N](f32->bf16) + bias)
// 256 threads (8 waves). Block tile 64x128, wave tile 32x32,
// K step 32, double-buffered LDS (1 barrier / K-step).
// ============================================================
template <int ACT, bool WF, bool WB>
__global__ void gemm_bf16_k(const unsigned short* __restrict__ A,
                            const float* __restrict__ Bw,
                            const float* __restrict__ bias,
                            float* __restrict__ outF,
                            unsigned short* __restrict__ outB,
                            int M, int N, int K)
{
    __shared__ __align__(16) unsigned short As[2][64][40];   // [m][k], row 80B
    __shared__ __align__(16) unsigned short Bt[2][128][40];  // [n][k] transposed, row 80B
    const int bm = blockIdx.y * 64, bn = blockIdx.x * 128;
    const int tid = threadIdx.x, lane = tid & 31, wave = tid >> 5;
    const int wm = (wave & 1) * 32, wn = (wave >> 1) * 32;
    const int l16 = lane & 15, half = lane >> 4;

    v8f acc00 = zero8(), acc01 = zero8(), acc10 = zero8(), acc11 = zero8();

    auto loadA = [&](int k0, int buf){
        const int r = tid >> 2, cq = (tid & 3) * 8;            // 64 rows x 4 chunks
        *(uint4*)(&As[buf][r][cq]) =
            *(const uint4*)(A + (size_t)(bm + r) * K + k0 + cq);
    };
    auto loadB = [&](int k0, int buf){
        for (int w = tid; w < 512; w += 256) {                 // 16 k-pairs x 32 n-quads
            const int k = (w >> 5) * 2, n0 = (w & 31) * 4;
            const float* src = Bw + (size_t)(k0 + k) * N + bn + n0;
            float4 f0 = *(const float4*)src;
            float4 f1 = *(const float4*)(src + N);
            *(unsigned*)(&Bt[buf][n0+0][k]) = (unsigned)f2bfu(f0.x) | ((unsigned)f2bfu(f1.x) << 16);
            *(unsigned*)(&Bt[buf][n0+1][k]) = (unsigned)f2bfu(f0.y) | ((unsigned)f2bfu(f1.y) << 16);
            *(unsigned*)(&Bt[buf][n0+2][k]) = (unsigned)f2bfu(f0.z) | ((unsigned)f2bfu(f1.z) << 16);
            *(unsigned*)(&Bt[buf][n0+3][k]) = (unsigned)f2bfu(f0.w) | ((unsigned)f2bfu(f1.w) << 16);
        }
    };

    loadA(0, 0); loadB(0, 0);
    __syncthreads();

    int buf = 0;
    for (int k0 = 0; k0 < K; k0 += 32) {
        const int nbuf = buf ^ 1;
        if (k0 + 32 < K) { loadA(k0 + 32, nbuf); loadB(k0 + 32, nbuf); }
        if (k0 + 64 < K) {                                      // L2 prefetch of K+2 tile
            __builtin_prefetch(A  + (size_t)(bm + (tid >> 2)) * K + k0 + 64, 0, 1);
            __builtin_prefetch(Bw + (size_t)(k0 + 64 + (tid >> 3)) * N + bn + (tid & 7) * 16, 0, 1);
        }

        v16u a0 = frag2(&As[buf][wm      + l16][0], half*8, 16 + half*8);
        v16u a1 = frag2(&As[buf][wm + 16 + l16][0], half*8, 16 + half*8);
        v16u b0 = frag2(&Bt[buf][wn      + l16][0], half*8, 16 + half*8);
        v16u b1 = frag2(&Bt[buf][wn + 16 + l16][0], half*8, 16 + half*8);

        acc00 = wmma_bf16(a0, b0, acc00);
        acc01 = wmma_bf16(a0, b1, acc01);
        acc10 = wmma_bf16(a1, b0, acc10);
        acc11 = wmma_bf16(a1, b1, acc11);

        __syncthreads();
        buf = nbuf;
    }

    const int col0 = bn + wn + l16, col1 = col0 + 16;
    const float bv0 = bias[col0], bv1 = bias[col1];
    auto emit = [&](v8f acc, int rowBase, int col, float bv){
        for (int r = 0; r < 8; r++) {
            const int row = rowBase + r + 8*half;
            float v = acc[r] + bv;
            if (ACT == 1) v = gelu_f(v);
            if (ACT == 2) v = fmaxf(v, 0.f);
            const size_t idx = (size_t)row * N + col;
            if (WF) outF[idx] = v;
            if (WB) outB[idx] = f2bfu(v);
        }
    };
    emit(acc00, bm + wm,      col0, bv0);
    emit(acc01, bm + wm,      col1, bv1);
    emit(acc10, bm + wm + 16, col0, bv0);
    emit(acc11, bm + wm + 16, col1, bv1);
}

// ============================================================
// Attention scores: per (b,h) block, S x S = Q(128x64) * K^T
// K^T fragment reads are contiguous since Ks is [s][d] = [n][k].
// ============================================================
__global__ void attn_scores_k(const unsigned short* __restrict__ qb,
                              const unsigned short* __restrict__ kb,
                              float* __restrict__ scores)
{
    const int bh = blockIdx.x, b = bh / NHn, h = bh % NHn;
    __shared__ __align__(16) unsigned short Qs[128][80];
    __shared__ __align__(16) unsigned short Ks[128][80];
    for (int w = threadIdx.x; w < 1024; w += 256) {
        const int s = w >> 3, dq = (w & 7) * 8;
        const size_t off = ((size_t)(b*Ssz + s)) * Hsz + h*DHn + dq;
        *(uint4*)(&Qs[s][dq]) = *(const uint4*)(qb + off);
        *(uint4*)(&Ks[s][dq]) = *(const uint4*)(kb + off);
    }
    __syncthreads();
    const int lane = threadIdx.x & 31, wave = threadIdx.x >> 5;
    const int l16 = lane & 15, half = lane >> 4;
    const int mBase = wave * 16;

    v8f acc[8];
    for (int n = 0; n < 8; n++) acc[n] = zero8();

    for (int kk = 0; kk < 64; kk += 32) {
        v16u a = frag2(&Qs[mBase + l16][0], kk + half*8, kk + 16 + half*8);
        for (int n = 0; n < 8; n++) {
            v16u bfrag = frag2(&Ks[n*16 + l16][0], kk + half*8, kk + 16 + half*8);
            acc[n] = wmma_bf16(a, bfrag, acc[n]);
        }
    }
    for (int n = 0; n < 8; n++)
        for (int r = 0; r < 8; r++) {
            const int row = mBase + r + 8*half, col = n*16 + l16;
            scores[((size_t)bh * 128 + row) * 128 + col] = acc[n][r];
        }
}

// ============================================================
// Softmax with scale + additive mask; one 128-thread block / row.
// ============================================================
__global__ void softmax_k(const float* __restrict__ scores,
                          const int* __restrict__ amask,
                          unsigned short* __restrict__ probs)
{
    const int row = blockIdx.x;          // bh*128 + q
    const int b = (row >> 7) / NHn;
    const int t = threadIdx.x;
    __shared__ float red[128];
    float v = scores[(size_t)row*128 + t] * INV_SQRT_DH
            + (amask[b*Ssz + t] > 0 ? 0.f : -1e9f);
    red[t] = v; __syncthreads();
    for (int s = 64; s > 0; s >>= 1) { if (t < s) red[t] = fmaxf(red[t], red[t+s]); __syncthreads(); }
    float mx = red[0]; __syncthreads();
    float e = __expf(v - mx);
    red[t] = e; __syncthreads();
    for (int s = 64; s > 0; s >>= 1) { if (t < s) red[t] += red[t+s]; __syncthreads(); }
    probs[(size_t)row*128 + t] = f2bfu(e / red[0]);
}

// ============================================================
// Context: per (b,h), ctx(128x64) = probs(128x128) * V(128x64)
// V is transposed into LDS at fill time -> contiguous B frags.
// ============================================================
__global__ void attn_ctx_k(const unsigned short* __restrict__ probs,
                           const unsigned short* __restrict__ vb,
                           unsigned short* __restrict__ ctxb)
{
    const int bh = blockIdx.x, b = bh / NHn, h = bh % NHn;
    __shared__ __align__(16) unsigned short Ps[128][136];   // [q][k], row 272B
    __shared__ __align__(16) unsigned short Vt[64][136];    // [d][s] transposed
    for (int w = threadIdx.x; w < 2048; w += 256) {
        const int r = w >> 4, cq = (w & 15) * 8;
        *(uint4*)(&Ps[r][cq]) = *(const uint4*)(probs + ((size_t)bh*128 + r)*128 + cq);
    }
    for (int w = threadIdx.x; w < 1024; w += 256) {
        const int s = w >> 3, dq = (w & 7) * 8;
        uint4 val = *(const uint4*)(vb + ((size_t)(b*Ssz + s)) * Hsz + h*DHn + dq);
        const unsigned short* p = (const unsigned short*)&val;
        for (int e = 0; e < 8; e++) Vt[dq + e][s] = p[e];
    }
    __syncthreads();
    const int lane = threadIdx.x & 31, wave = threadIdx.x >> 5;
    const int l16 = lane & 15, half = lane >> 4;
    const int mBase = wave * 16;

    v8f acc[4];
    for (int n = 0; n < 4; n++) acc[n] = zero8();

    for (int kk = 0; kk < 128; kk += 32) {
        v16u a = frag2(&Ps[mBase + l16][0], kk + half*8, kk + 16 + half*8);
        for (int n = 0; n < 4; n++) {
            v16u bfrag = frag2(&Vt[n*16 + l16][0], kk + half*8, kk + 16 + half*8);
            acc[n] = wmma_bf16(a, bfrag, acc[n]);
        }
    }
    for (int n = 0; n < 4; n++)
        for (int r = 0; r < 8; r++) {
            const int row = mBase + r + 8*half, col = n*16 + l16;
            ctxb[((size_t)(b*Ssz + row)) * Hsz + h*DHn + col] = f2bfu(acc[n][r]);
        }
}

// ============================================================
// Embedding + LayerNorm (one block per token)
// ============================================================
__global__ void embed_ln_k(const int* __restrict__ ids,
                           const float* __restrict__ wemb,
                           const float* __restrict__ pemb,
                           const float* __restrict__ gs, const float* __restrict__ gb,
                           float* __restrict__ x, unsigned short* __restrict__ xb)
{
    const int tok = blockIdx.x, s = tok % Ssz;
    const int id = ids[tok];
    __shared__ float red[256];
    float local = 0.f;
    for (int i = threadIdx.x; i < Hsz; i += 256) {
        float v = wemb[(size_t)id*Hsz + i] + pemb[(size_t)s*Hsz + i];
        x[(size_t)tok*Hsz + i] = v;
        local += v;
    }
    red[threadIdx.x] = local; __syncthreads();
    for (int t = 128; t > 0; t >>= 1) { if (threadIdx.x < t) red[threadIdx.x] += red[threadIdx.x+t]; __syncthreads(); }
    float mean = red[0] / Hsz; __syncthreads();
    local = 0.f;
    for (int i = threadIdx.x; i < Hsz; i += 256) {
        float d = x[(size_t)tok*Hsz + i] - mean; local += d*d;
    }
    red[threadIdx.x] = local; __syncthreads();
    for (int t = 128; t > 0; t >>= 1) { if (threadIdx.x < t) red[threadIdx.x] += red[threadIdx.x+t]; __syncthreads(); }
    float inv = rsqrtf(red[0] / Hsz + LN_EPS);
    for (int i = threadIdx.x; i < Hsz; i += 256) {
        float v = (x[(size_t)tok*Hsz + i] - mean) * inv * gs[i] + gb[i];
        x[(size_t)tok*Hsz + i]  = v;
        xb[(size_t)tok*Hsz + i] = f2bfu(v);
    }
}

// ============================================================
// Residual add + LayerNorm (in-place on x), writes bf16 copy
// ============================================================
__global__ void add_ln_k(const float* __restrict__ delta,
                         const float* __restrict__ gs, const float* __restrict__ gb,
                         float* __restrict__ x, unsigned short* __restrict__ xb)
{
    const int tok = blockIdx.x;
    __shared__ float red[256];
    float local = 0.f;
    for (int i = threadIdx.x; i < Hsz; i += 256) {
        float v = x[(size_t)tok*Hsz + i] + delta[(size_t)tok*Hsz + i];
        x[(size_t)tok*Hsz + i] = v;
        local += v;
    }
    red[threadIdx.x] = local; __syncthreads();
    for (int t = 128; t > 0; t >>= 1) { if (threadIdx.x < t) red[threadIdx.x] += red[threadIdx.x+t]; __syncthreads(); }
    float mean = red[0] / Hsz; __syncthreads();
    local = 0.f;
    for (int i = threadIdx.x; i < Hsz; i += 256) {
        float d = x[(size_t)tok*Hsz + i] - mean; local += d*d;
    }
    red[threadIdx.x] = local; __syncthreads();
    for (int t = 128; t > 0; t >>= 1) { if (threadIdx.x < t) red[threadIdx.x] += red[threadIdx.x+t]; __syncthreads(); }
    float inv = rsqrtf(red[0] / Hsz + LN_EPS);
    for (int i = threadIdx.x; i < Hsz; i += 256) {
        float v = (x[(size_t)tok*Hsz + i] - mean) * inv * gs[i] + gb[i];
        x[(size_t)tok*Hsz + i]  = v;
        xb[(size_t)tok*Hsz + i] = f2bfu(v);
    }
}

// ---------------- small head / loss kernels ----------------
__global__ void relu_cvt_k(const float* __restrict__ x, unsigned short* __restrict__ xb, int n){
    int i = blockIdx.x * 256 + threadIdx.x;
    if (i < n) xb[i] = f2bfu(fmaxf(x[i], 0.f));
}

__global__ void zero_scalar_k(float* p){ if (threadIdx.x == 0) p[0] = 0.f; }

__global__ void intent_h1_k(const float* __restrict__ x, const float* __restrict__ iW1,
                            const float* __restrict__ ib1, float* __restrict__ h1){
    int b = blockIdx.x, j = threadIdx.x;   // 64 threads
    const float* xr = x + (size_t)b * Ssz * Hsz;   // token 0
    float acc = ib1[j];
    for (int k = 0; k < Hsz; k++) acc += fmaxf(xr[k], 0.f) * iW1[k*64 + j];
    h1[b*64 + j] = fmaxf(acc, 0.f);
}

__global__ void intent_h2_k(const float* __restrict__ h1, const float* __restrict__ iW2,
                            const float* __restrict__ ib2, const int* __restrict__ target,
                            float* __restrict__ out_pred, float* __restrict__ ilsum){
    int b = blockIdx.x, c = threadIdx.x;   // 32 threads
    __shared__ float lg[32];
    float acc = -1e30f;
    if (c < NINT) {
        acc = ib2[c];
        for (int j = 0; j < 64; j++) acc += h1[b*64 + j] * iW2[j*NINT + c];
    }
    lg[c] = acc; __syncthreads();
    if (c == 0) {
        float mx = -1e30f; int bi = 0;
        for (int i = 0; i < NINT; i++) if (lg[i] > mx) { mx = lg[i]; bi = i; }
        float sum = 0.f;
        for (int i = 0; i < NINT; i++) sum += __expf(lg[i] - mx);
        float loss = mx + __logf(sum) - lg[target[b]];
        atomicAdd(ilsum, loss);
        out_pred[b] = (float)bi;
    }
}

__global__ void slot_h2_k(const unsigned short* __restrict__ sh, const float* __restrict__ sW2,
                          const float* __restrict__ sb2, float* __restrict__ logits){
    int row = blockIdx.x, c = threadIdx.x;
    if (c >= NTAG) return;
    float acc = sb2[c];
    for (int j = 0; j < 256; j++) acc += bfu2f(sh[(size_t)row*256 + j]) * sW2[j*NTAG + c];
    logits[(size_t)row*NTAG + c] = acc;
}

// ---------------- CRF kernels ----------------
__global__ void crf_num_k(const float* __restrict__ emis, const int* __restrict__ tags,
                          const int* __restrict__ mask, const float* __restrict__ cstart,
                          const float* __restrict__ cend, const float* __restrict__ ctrans,
                          float* __restrict__ num){
    int b = threadIdx.x; if (b >= Bsz) return;
    int t0 = tags[b*Ssz];
    float n = cstart[t0] + emis[((size_t)(b*Ssz)) * NTAG + t0];
    int mcount = mask[b*Ssz];
    for (int t = 1; t < Ssz; t++) {
        int tt = tags[b*Ssz + t], tp = tags[b*Ssz + t - 1];
        float fm = mask[b*Ssz + t] ? 1.f : 0.f;
        n += (ctrans[tp*NTAG + tt] + emis[((size_t)(b*Ssz + t)) * NTAG + tt]) * fm;
        mcount += mask[b*Ssz + t];
    }
    n += cend[tags[b*Ssz + (mcount - 1)]];
    num[b] = n;
}

__global__ void crf_forward_k(const float* __restrict__ emis, const int* __restrict__ mask,
                              const float* __restrict__ cstart, const float* __restrict__ cend,
                              const float* __restrict__ ctrans, float* __restrict__ denom){
    int b = blockIdx.x, j = threadIdx.x;   // 128 threads
    __shared__ float alpha[NTAG];
    __shared__ float red[128];
    if (j < NTAG) alpha[j] = cstart[j] + emis[((size_t)(b*Ssz)) * NTAG + j];
    __syncthreads();
    for (int t = 1; t < Ssz; t++) {
        float nxt = -1e30f;
        if (j < NTAG) {
            float mx = -1e30f;
            for (int i = 0; i < NTAG; i++) mx = fmaxf(mx, alpha[i] + ctrans[i*NTAG + j]);
            float sum = 0.f;
            for (int i = 0; i < NTAG; i++) sum += __expf(alpha[i] + ctrans[i*NTAG + j] - mx);
            nxt = mx + __logf(sum) + emis[((size_t)(b*Ssz + t)) * NTAG + j];
        }
        __syncthreads();
        if (j < NTAG && mask[b*Ssz + t]) alpha[j] = nxt;
        __syncthreads();
    }
    float v = (j < NTAG) ? alpha[j] + cend[j] : -1e30f;
    red[j] = v; __syncthreads();
    for (int s = 64; s > 0; s >>= 1) { if (j < s) red[j] = fmaxf(red[j], red[j+s]); __syncthreads(); }
    float mx = red[0]; __syncthreads();
    red[j] = __expf(v - mx); __syncthreads();
    for (int s = 64; s > 0; s >>= 1) { if (j < s) red[j] += red[j+s]; __syncthreads(); }
    if (j == 0) denom[b] = mx + __logf(red[0]);
}

__global__ void viterbi_k(const float* __restrict__ emis, const int* __restrict__ mask,
                          const float* __restrict__ cstart, const float* __restrict__ cend,
                          const float* __restrict__ ctrans, int* __restrict__ bp,
                          float* __restrict__ out_pred){
    int b = blockIdx.x, j = threadIdx.x;   // 128 threads
    __shared__ float alpha[NTAG];
    __shared__ float red[128];
    __shared__ int   redi[128];
    if (j < NTAG) alpha[j] = cstart[j] + emis[((size_t)(b*Ssz)) * NTAG + j];
    __syncthreads();
    for (int t = 1; t < Ssz; t++) {
        float best = -1e30f; int bi = (j < NTAG) ? j : 0;
        if (j < NTAG) {
            for (int i = 0; i < NTAG; i++) {
                float sc = alpha[i] + ctrans[i*NTAG + j];
                if (sc > best) { best = sc; bi = i; }
            }
        }
        __syncthreads();
        if (j < NTAG) {
            size_t bidx = ((size_t)b*(Ssz-1) + (t-1)) * NTAG + j;
            if (mask[b*Ssz + t]) {
                bp[bidx] = bi;
                alpha[j] = best + emis[((size_t)(b*Ssz + t)) * NTAG + j];
            } else {
                bp[bidx] = j;
            }
        }
        __syncthreads();
    }
    float v = (j < NTAG) ? alpha[j] + cend[j] : -1e30f;
    red[j] = v; redi[j] = (j < NTAG) ? j : 0; __syncthreads();
    for (int s = 64; s > 0; s >>= 1) {
        if (j < s && red[j+s] > red[j]) { red[j] = red[j+s]; redi[j] = redi[j+s]; }
        __syncthreads();
    }
    if (j == 0) {
        int tag = redi[0];
        out_pred[b*Ssz + (Ssz-1)] = (float)tag;
        for (int t = Ssz - 2; t >= 0; t--) {
            tag = bp[((size_t)b*(Ssz-1) + t) * NTAG + tag];
            out_pred[b*Ssz + t] = (float)tag;
        }
    }
}

__global__ void finalize_k(const float* __restrict__ num, const float* __restrict__ denom,
                           const float* __restrict__ ilsum, const float* __restrict__ log_vars,
                           float* __restrict__ out){
    if (threadIdx.x == 0) {
        float sl = 0.f;
        for (int b = 0; b < Bsz; b++) sl += num[b] - denom[b];
        sl = -sl;
        float p1 = __expf(-log_vars[0]), p2 = __expf(-log_vars[1]);
        out[0] = p1 * ilsum[0] + Bsz * log_vars[0] + p2 * sl + log_vars[1];
    }
}

// ============================================================
// host orchestration
// ============================================================
extern "C" void kernel_launch(void* const* d_in, const int* in_sizes, int n_in,
                              void* d_out, int out_size, void* d_ws, size_t ws_size,
                              hipStream_t stream)
{
    (void)in_sizes; (void)n_in; (void)out_size; (void)ws_size;
    const int*   ids      = (const int*)  d_in[0];
    const int*   amask    = (const int*)  d_in[1];
    const int*   itgt     = (const int*)  d_in[2];
    const int*   stgt     = (const int*)  d_in[3];
    const int*   smask    = (const int*)  d_in[4];
    const float* wemb     = (const float*)d_in[5];
    const float* pemb     = (const float*)d_in[6];
    const float* eln_s    = (const float*)d_in[7];
    const float* eln_b    = (const float*)d_in[8];
    const float* Wq       = (const float*)d_in[9];
    const float* bq       = (const float*)d_in[10];
    const float* Wk       = (const float*)d_in[11];
    const float* bk       = (const float*)d_in[12];
    const float* Wv       = (const float*)d_in[13];
    const float* bv       = (const float*)d_in[14];
    const float* Wo       = (const float*)d_in[15];
    const float* bo       = (const float*)d_in[16];
    const float* saln_s   = (const float*)d_in[17];
    const float* saln_b   = (const float*)d_in[18];
    const float* W1f      = (const float*)d_in[19];
    const float* b1f      = (const float*)d_in[20];
    const float* W2f      = (const float*)d_in[21];
    const float* b2f      = (const float*)d_in[22];
    const float* oln_s    = (const float*)d_in[23];
    const float* oln_b    = (const float*)d_in[24];
    const float* iW1      = (const float*)d_in[25];
    const float* ib1      = (const float*)d_in[26];
    const float* iW2      = (const float*)d_in[27];
    const float* ib2      = (const float*)d_in[28];
    const float* sW1      = (const float*)d_in[29];
    const float* sb1      = (const float*)d_in[30];
    const float* sW2      = (const float*)d_in[31];
    const float* sb2      = (const float*)d_in[32];
    const float* cstart   = (const float*)d_in[33];
    const float* cend     = (const float*)d_in[34];
    const float* ctrans   = (const float*)d_in[35];
    const float* logvars  = (const float*)d_in[36];

    float* out = (float*)d_out;             // [0]=loss, [1..8192]=slots_pred, [8193..]=intent_pred
    float* out_slots  = out + 1;
    float* out_intent = out + 1 + Bsz*Ssz;

    // ---- workspace bump allocator (~165 MB with aliasing) ----
    char* w = (char*)d_ws;
    auto alloc = [&](size_t bytes) -> void* {
        void* p = (void*)w;
        w += (bytes + 255) & ~(size_t)255;
        return p;
    };
    float*          xf     = (float*)         alloc((size_t)TOKS*Hsz*4);
    unsigned short* xb     = (unsigned short*)alloc((size_t)TOKS*Hsz*2);
    unsigned short* qb     = (unsigned short*)alloc((size_t)TOKS*Hsz*2);   // q / ctx / relu(x)
    unsigned short* kbuf   = (unsigned short*)alloc((size_t)TOKS*Hsz*2);
    unsigned short* vbuf   = (unsigned short*)alloc((size_t)TOKS*Hsz*2);
    float*          scores = (float*)         alloc((size_t)Bsz*NHn*128*128*4);
    unsigned short* hb     = (unsigned short*)scores;                       // alias (same byte count)
    float*          tmpf   = (float*)         alloc((size_t)TOKS*Hsz*4);
    unsigned short* probs  = (unsigned short*)tmpf;                         // alias (same byte count)
    unsigned short* shb    = (unsigned short*)alloc((size_t)TOKS*256*2);
    float*          slogit = (float*)         alloc((size_t)TOKS*NTAG*4);
    float*          h1     = (float*)         alloc((size_t)Bsz*64*4);
    float*          num    = (float*)         alloc((size_t)Bsz*4);
    float*          denom  = (float*)         alloc((size_t)Bsz*4);
    float*          ilsum  = (float*)         alloc(256);
    int*            bp     = (int*)           alloc((size_t)Bsz*(Ssz-1)*NTAG*4);

    embed_ln_k<<<TOKS, 256, 0, stream>>>(ids, wemb, pemb, eln_s, eln_b, xf, xb);

    for (int l = 0; l < Lnum; l++) {
        const float* Wq_l = Wq + (size_t)l*Hsz*Hsz;  const float* bq_l = bq + l*Hsz;
        const float* Wk_l = Wk + (size_t)l*Hsz*Hsz;  const float* bk_l = bk + l*Hsz;
        const float* Wv_l = Wv + (size_t)l*Hsz*Hsz;  const float* bv_l = bv + l*Hsz;
        const float* Wo_l = Wo + (size_t)l*Hsz*Hsz;  const float* bo_l = bo + l*Hsz;
        const float* W1_l = W1f + (size_t)l*Hsz*FFn; const float* b1_l = b1f + l*FFn;
        const float* W2_l = W2f + (size_t)l*FFn*Hsz; const float* b2_l = b2f + l*Hsz;

        dim3 gH(Hsz/128, TOKS/64);   // N=768
        gemm_bf16_k<0,false,true><<<gH, 256, 0, stream>>>(xb, Wq_l, bq_l, nullptr, qb,   TOKS, Hsz, Hsz);
        gemm_bf16_k<0,false,true><<<gH, 256, 0, stream>>>(xb, Wk_l, bk_l, nullptr, kbuf, TOKS, Hsz, Hsz);
        gemm_bf16_k<0,false,true><<<gH, 256, 0, stream>>>(xb, Wv_l, bv_l, nullptr, vbuf, TOKS, Hsz, Hsz);

        attn_scores_k<<<Bsz*NHn, 256, 0, stream>>>(qb, kbuf, scores);
        softmax_k<<<Bsz*NHn*Ssz, 128, 0, stream>>>(scores, amask, probs);
        attn_ctx_k<<<Bsz*NHn, 256, 0, stream>>>(probs, vbuf, qb /* ctx reuses q buffer */);

        gemm_bf16_k<0,true,false><<<gH, 256, 0, stream>>>(qb, Wo_l, bo_l, tmpf, nullptr, TOKS, Hsz, Hsz);
        add_ln_k<<<TOKS, 256, 0, stream>>>(tmpf, saln_s + l*Hsz, saln_b + l*Hsz, xf, xb);

        dim3 gF1(FFn/128, TOKS/64);  // N=3072
        gemm_bf16_k<1,false,true><<<gF1, 256, 0, stream>>>(xb, W1_l, b1_l, nullptr, hb, TOKS, FFn, Hsz);
        gemm_bf16_k<0,true,false><<<gH, 256, 0, stream>>>(hb, W2_l, b2_l, tmpf, nullptr, TOKS, Hsz, FFn);
        add_ln_k<<<TOKS, 256, 0, stream>>>(tmpf, oln_s + l*Hsz, oln_b + l*Hsz, xf, xb);
    }

    // ---- slot head: relu(x) @ sW1 -> relu -> @ sW2 ----
    relu_cvt_k<<<(TOKS*Hsz + 255)/256, 256, 0, stream>>>(xf, qb, TOKS*Hsz);
    dim3 gS(256/128, TOKS/64);
    gemm_bf16_k<2,false,true><<<gS, 256, 0, stream>>>(qb, sW1, sb1, nullptr, shb, TOKS, 256, Hsz);
    slot_h2_k<<<TOKS, 128, 0, stream>>>(shb, sW2, sb2, slogit);

    // ---- intent head + loss ----
    zero_scalar_k<<<1, 1, 0, stream>>>(ilsum);
    intent_h1_k<<<Bsz, 64, 0, stream>>>(xf, iW1, ib1, h1);
    intent_h2_k<<<Bsz, 32, 0, stream>>>(h1, iW2, ib2, itgt, out_intent, ilsum);

    // ---- CRF log-likelihood + Viterbi ----
    crf_num_k<<<1, 64, 0, stream>>>(slogit, stgt, smask, cstart, cend, ctrans, num);
    crf_forward_k<<<Bsz, 128, 0, stream>>>(slogit, smask, cstart, cend, ctrans, denom);
    viterbi_k<<<Bsz, 128, 0, stream>>>(slogit, smask, cstart, cend, ctrans, bp, out_slots);

    finalize_k<<<1, 1, 0, stream>>>(num, denom, ilsum, logvars, out);
}
